// AttnDecoderRNN_12275016532425
// MI455X (gfx1250) — matrix operand order repaired and checked by reading, not compiled, not run on hardware
//
#include <hip/hip_runtime.h>
#include <stdint.h>

// ---------------- problem constants ----------------
constexpr int B   = 32;
constexpr int S   = 256;
constexpr int H   = 512;
constexpr int ENC = 512;
constexpr int V   = 32000;
constexpr int L   = 64;
constexpr int SOS = 1;
constexpr int KX  = ENC + H;   // 1024, GRU input width
constexpr int G3  = 3 * H;     // 1536

typedef unsigned short bfu16;  // raw bf16 bits
typedef __attribute__((ext_vector_type(16))) __bf16 v16bf;
typedef __attribute__((ext_vector_type(8)))  float  v8f;

// fp32 -> bf16 (round-to-nearest-even)
static __device__ __forceinline__ bfu16 f2bf(float f) {
    unsigned int u = __float_as_uint(f);
    u += 0x7FFFu + ((u >> 16) & 1u);
    return (bfu16)(u >> 16);
}

struct U8x4 { uint4 lo, hi; };
union FragCvt { U8x4 u; v16bf v; };

// A fragment: 16x32 bf16, rows m0..m0+15, cols k0..k0+31 from row-major [., ld]
static __device__ __forceinline__ v16bf frag_A(const bfu16* base, int ld, int m0, int k0) {
    int lane = threadIdx.x & 31;
    int m    = m0 + (lane & 15);
    int hi   = lane >> 4;                       // half-wave selector
    const bfu16* p = base + (size_t)m * ld + k0 + hi * 8;
    FragCvt c;
    c.u.lo = *(const uint4*)p;                  // K = k0+hi*8   .. +7
    c.u.hi = *(const uint4*)(p + 16);           // K = k0+16+hi*8.. +7
    return c.v;
}

// B fragment: 32x16 (K x N) where B[k][n] = W[n0+n][k0+k], W row-major (N,K)
static __device__ __forceinline__ v16bf frag_B(const bfu16* base, int ld, int n0, int k0) {
    int lane = threadIdx.x & 31;
    int n    = n0 + (lane & 15);
    int kh   = (lane >> 4) * 8;
    const bfu16* p = base + (size_t)n * ld + k0 + kh;
    FragCvt c;
    c.u.lo = *(const uint4*)p;                  // K = k0+kh .. +7, then k0+16+kh .. +7
    c.u.hi = *(const uint4*)(p + 16);
    return c.v;
}

static __device__ __forceinline__ v8f wmma_bf16(v16bf a, v16bf b, v8f c) {
    return __builtin_amdgcn_wmma_f32_16x16x32_bf16(
        /*neg_a=*/false, a, /*neg_b=*/false, b,
        /*c_mod=*/(short)0, c, /*reuse_a=*/false, /*reuse_b=*/false);
}

// gfx1250 async copy global->LDS (16B per lane), tracked by ASYNCcnt
static __device__ __forceinline__ void async_ld16(uint32_t lds_off, const void* g) {
    asm volatile("global_load_async_to_lds_b128 %0, %1, off"
                 :: "v"(lds_off), "v"(g) : "memory");
}
static __device__ __forceinline__ void wait_async0() {
    asm volatile("s_wait_asynccnt 0x0" ::: "memory");
}

// ---------------- prep kernels ----------------
__global__ void f2bf_kernel(const float* __restrict__ src, bfu16* __restrict__ dst, int n) {
    int i = blockIdx.x * blockDim.x + threadIdx.x;
    int stride = gridDim.x * blockDim.x;
    for (; i < n; i += stride) dst[i] = f2bf(src[i]);
}

// h0 = encoder_hidden[0] @ W_ht^T + b_ht   (B x H), also bf16 copy
__global__ void h0_kernel(const float* __restrict__ ehid, const float* __restrict__ Wht,
                          const float* __restrict__ bht, float* __restrict__ hf,
                          bfu16* __restrict__ hb) {
    int b = blockIdx.x;
    for (int j = threadIdx.x; j < H; j += blockDim.x) {
        float s = bht[j];
        const float* x = ehid + (size_t)b * ENC;
        const float* w = Wht + (size_t)j * ENC;
        for (int k = 0; k < ENC; ++k) s += x[k] * w[k];
        hf[b * H + j] = s;
        hb[b * H + j] = f2bf(s);
    }
}

// Uk = enc @ Ua^T + bUa : (B*S=8192) x 512, K=512.  Each wave: 2 m-tiles, B reused.
__global__ void uk_kernel(const bfu16* __restrict__ enc_bf, const bfu16* __restrict__ Ua_bf,
                          const float* __restrict__ bUa, float* __restrict__ Uk) {
    int wave = threadIdx.x >> 5;
    int wid  = blockIdx.x * 8 + wave;      // 0 .. 8191
    int nt   = wid & 31;                   // 32 n-tiles over H
    int mp   = wid >> 5;                   // 256 m-tile pairs over B*S
    int m0 = mp * 32, n0 = nt * 16;
    v8f a0 = {}, a1 = {};
    for (int k0 = 0; k0 < ENC; k0 += 32) {
        v16bf bf_ = frag_B(Ua_bf, ENC, n0, k0);
        a0 = wmma_bf16(frag_A(enc_bf, ENC, m0,      k0), bf_, a0);
        a1 = wmma_bf16(frag_A(enc_bf, ENC, m0 + 16, k0), bf_, a1);
    }
    int lane = threadIdx.x & 31;
    int n = n0 + (lane & 15);
    float bb = bUa[n];
    int mm = m0 + (lane >> 4) * 8;
    for (int v = 0; v < 8; ++v) {
        Uk[(size_t)(mm + v) * H + n]      = a0[v] + bb;
        Uk[(size_t)(mm + 16 + v) * H + n] = a1[v] + bb;
    }
}

// ---------------- per-step kernels ----------------
// Attention: embedding gather, q = h@Wa^T+ba, additive scores, softmax,
// context, writes x = [e, ctx] in bf16, attention weights to output.
__global__ void attn_kernel(int t, const int* __restrict__ tgt, const float* __restrict__ emb,
                            const float* __restrict__ Wa, const float* __restrict__ ba,
                            const float* __restrict__ Va, const float* __restrict__ bVa,
                            const float* __restrict__ Uk, const float* __restrict__ enc,
                            const float* __restrict__ h, bfu16* __restrict__ x_bf,
                            float* __restrict__ attn_out) {
    __shared__ float q[H];
    __shared__ float sw[S];
    __shared__ float red[256];
    int b = blockIdx.x, tid = threadIdx.x;
    int tok = (t == 0) ? SOS : tgt[b * L + (t - 1)];

    for (int j = tid; j < H; j += 256)              // embedding -> x[:, 0:H]
        x_bf[b * KX + j] = f2bf(emb[(size_t)tok * H + j]);

    const float* hr = h + (size_t)b * H;
    for (int j = tid; j < H; j += 256) {            // q[j]
        float s = ba[j];
        const float* w = Wa + (size_t)j * H;
        for (int k = 0; k < H; ++k) s += hr[k] * w[k];
        q[j] = s;
    }
    __syncthreads();

    // one thread per source position (S == 256 == blockDim)
    float sc = bVa[0];
    const float* uk = Uk + (size_t)(b * S + tid) * H;
    for (int j = 0; j < H; ++j) sc += Va[j] * tanhf(q[j] + uk[j]);

    red[tid] = sc; __syncthreads();
    for (int o = 128; o > 0; o >>= 1) { if (tid < o) red[tid] = fmaxf(red[tid], red[tid + o]); __syncthreads(); }
    float mx = red[0]; __syncthreads();
    float ex = __expf(sc - mx);
    red[tid] = ex; __syncthreads();
    for (int o = 128; o > 0; o >>= 1) { if (tid < o) red[tid] += red[tid + o]; __syncthreads(); }
    float wsft = ex / red[0];
    sw[tid] = wsft;
    attn_out[(size_t)(b * L + t) * S + tid] = wsft;
    __syncthreads();

    for (int e0 = tid; e0 < ENC; e0 += 256) {       // ctx -> x[:, H:H+ENC]
        float c = 0.f;
        const float* ep = enc + (size_t)b * S * ENC + e0;
        for (int s2 = 0; s2 < S; ++s2) c += sw[s2] * ep[(size_t)s2 * ENC];
        x_bf[b * KX + H + e0] = f2bf(c);
    }
}

// GRU gates via WMMA + fused pointwise update. 8 WGs, each owns 64 hidden cols.
// h_bf staged in LDS via async copy; each job computes both m-tiles (B reuse).
__global__ void gru_kernel(const bfu16* __restrict__ x_bf, const bfu16* __restrict__ hbf,
                           const float* __restrict__ hf, const bfu16* __restrict__ Wih,
                           const bfu16* __restrict__ Whh, const float* __restrict__ bih,
                           const float* __restrict__ bhh, float* __restrict__ hfo,
                           bfu16* __restrict__ hbo) {
    __shared__ __align__(16) bfu16 sH[B * H];                 // 32 KB
    __shared__ float sR[B][64], sZ[B][64], sGi[B][64], sGh[B][64];  // 32 KB
    int wg   = blockIdx.x;           // 0..7
    int tid  = threadIdx.x;
    int wave = tid >> 5;
    int lane = tid & 31;

    // stage h (bf16) into LDS: 2048 x 16B chunks
    {
        uint32_t base = (uint32_t)(uintptr_t)&sH[0];
        const char* g = (const char*)hbf;
        for (int i = tid; i < (B * H * 2) / 16; i += 256)
            async_ld16(base + i * 16, g + i * 16);
        wait_async0();
    }
    __syncthreads();

    // 16 jobs: grp 0:r(gi+gh) 1:z(gi+gh) 2:gi_n 3:gh_n, 4 n-tiles each
    for (int job = wave; job < 16; job += 8) {
        int grp = job >> 2;
        int ntile = job & 3;
        int nl0 = ntile * 16;
        v8f a0 = {}, a1 = {};
        if (grp <= 1) {
            int ng = grp * H + wg * 64 + nl0;
            for (int k0 = 0; k0 < KX; k0 += 32) {
                v16bf bf_ = frag_B(Wih, KX, ng, k0);
                a0 = wmma_bf16(frag_A(x_bf, KX, 0,  k0), bf_, a0);
                a1 = wmma_bf16(frag_A(x_bf, KX, 16, k0), bf_, a1);
            }
            for (int k0 = 0; k0 < H; k0 += 32) {
                v16bf bf_ = frag_B(Whh, H, ng, k0);
                a0 = wmma_bf16(frag_A(sH, H, 0,  k0), bf_, a0);
                a1 = wmma_bf16(frag_A(sH, H, 16, k0), bf_, a1);
            }
        } else if (grp == 2) {
            int ng = 2 * H + wg * 64 + nl0;
            for (int k0 = 0; k0 < KX; k0 += 32) {
                v16bf bf_ = frag_B(Wih, KX, ng, k0);
                a0 = wmma_bf16(frag_A(x_bf, KX, 0,  k0), bf_, a0);
                a1 = wmma_bf16(frag_A(x_bf, KX, 16, k0), bf_, a1);
            }
        } else {
            int ng = 2 * H + wg * 64 + nl0;
            for (int k0 = 0; k0 < H; k0 += 32) {
                v16bf bf_ = frag_B(Whh, H, ng, k0);
                a0 = wmma_bf16(frag_A(sH, H, 0,  k0), bf_, a0);
                a1 = wmma_bf16(frag_A(sH, H, 16, k0), bf_, a1);
            }
        }
        float (*dst)[64] = (grp == 0) ? sR : (grp == 1) ? sZ : (grp == 2) ? sGi : sGh;
        int mmb = (lane >> 4) * 8;
        int nn  = nl0 + (lane & 15);
        for (int v = 0; v < 8; ++v) {
            dst[mmb + v][nn]      = a0[v];
            dst[16 + mmb + v][nn] = a1[v];
        }
    }
    __syncthreads();

    for (int idx = tid; idx < B * 64; idx += 256) {
        int m = idx >> 6, jl = idx & 63;
        int j = wg * 64 + jl;
        float r  = 1.f / (1.f + __expf(-(sR[m][jl] + bih[j] + bhh[j])));
        float z  = 1.f / (1.f + __expf(-(sZ[m][jl] + bih[H + j] + bhh[H + j])));
        float n  = tanhf(sGi[m][jl] + bih[2 * H + j] + r * (sGh[m][jl] + bhh[2 * H + j]));
        float hn = (1.f - z) * n + z * hf[m * H + j];
        hfo[m * H + j] = hn;
        hbo[m * H + j] = f2bf(hn);
    }
}

// logits = h_new @ W_out^T + b_out : 32 x 32000, K=512.
// h staged in LDS (async), each wave: one n-tile, both m-tiles (B reuse).
__global__ void logits_kernel(const bfu16* __restrict__ hbf, const bfu16* __restrict__ Wout,
                              const float* __restrict__ bout, float* __restrict__ logits) {
    __shared__ __align__(16) bfu16 sA[B * H];    // 32 KB
    int tid = threadIdx.x;
    {
        uint32_t base = (uint32_t)(uintptr_t)&sA[0];
        const char* g = (const char*)hbf;
        for (int i = tid; i < (B * H * 2) / 16; i += 256)
            async_ld16(base + i * 16, g + i * 16);
        wait_async0();
    }
    __syncthreads();

    int wave = tid >> 5, lane = tid & 31;
    int nt = blockIdx.x * 8 + wave;     // 0..1999
    int n0 = nt * 16;
    v8f a0 = {}, a1 = {};
    for (int k0 = 0; k0 < H; k0 += 32) {
        v16bf bf_ = frag_B(Wout, H, n0, k0);
        a0 = wmma_bf16(frag_A(sA, H, 0,  k0), bf_, a0);
        a1 = wmma_bf16(frag_A(sA, H, 16, k0), bf_, a1);
    }
    int n = n0 + (lane & 15);
    float bb = bout[n];
    int mm = (lane >> 4) * 8;
    for (int v = 0; v < 8; ++v) {
        logits[(size_t)(mm + v) * V + n]      = a0[v] + bb;
        logits[(size_t)(mm + 16 + v) * V + n] = a1[v] + bb;
    }
}

// row-wise log_softmax over V, write to (B, L, V) output slice for step t
__global__ void lsm_kernel(int t, const float* __restrict__ logits, float* __restrict__ out) {
    __shared__ float red[256];
    int b = blockIdx.x, tid = threadIdx.x;
    const float* row = logits + (size_t)b * V;
    float mx = -3.4e38f;
    for (int v = tid; v < V; v += 256) mx = fmaxf(mx, row[v]);
    red[tid] = mx; __syncthreads();
    for (int o = 128; o > 0; o >>= 1) { if (tid < o) red[tid] = fmaxf(red[tid], red[tid + o]); __syncthreads(); }
    mx = red[0]; __syncthreads();
    float s = 0.f;
    for (int v = tid; v < V; v += 256) s += __expf(row[v] - mx);
    red[tid] = s; __syncthreads();
    for (int o = 128; o > 0; o >>= 1) { if (tid < o) red[tid] += red[tid + o]; __syncthreads(); }
    float lse = mx + __logf(red[0]);
    float* orow = out + (size_t)(b * L + t) * V;
    for (int v = tid; v < V; v += 256) orow[v] = row[v] - lse;
}

__global__ void hfinal_kernel(const float* __restrict__ hf, float* __restrict__ out) {
    int i = blockIdx.x * blockDim.x + threadIdx.x;
    if (i < B * H) out[i] = hf[i];
}

// ---------------- host launcher ----------------
extern "C" void kernel_launch(void* const* d_in, const int* in_sizes, int n_in,
                              void* d_out, int out_size, void* d_ws, size_t ws_size,
                              hipStream_t stream) {
    (void)in_sizes; (void)n_in; (void)out_size; (void)ws_size;
    const float* enc  = (const float*)d_in[0];
    const float* ehid = (const float*)d_in[1];
    const int*   tgt  = (const int*)  d_in[2];
    const float* Wht  = (const float*)d_in[3];
    const float* bht  = (const float*)d_in[4];
    const float* emb  = (const float*)d_in[5];
    const float* Wa   = (const float*)d_in[6];
    const float* ba   = (const float*)d_in[7];
    const float* Ua   = (const float*)d_in[8];
    const float* bUa  = (const float*)d_in[9];
    const float* Va   = (const float*)d_in[10];
    const float* bVa  = (const float*)d_in[11];
    const float* Wih  = (const float*)d_in[12];
    const float* Whh  = (const float*)d_in[13];
    const float* bih  = (const float*)d_in[14];
    const float* bhh  = (const float*)d_in[15];
    const float* Wout = (const float*)d_in[16];
    const float* bout = (const float*)d_in[17];

    char* ws = (char*)d_ws;
    size_t off = 0;
    auto take = [&](size_t nbytes) -> char* {
        off = (off + 255) & ~(size_t)255;
        char* p = ws + off; off += nbytes; return p;
    };
    float* Uk      = (float*)take((size_t)B * S * H * 4);
    bfu16* enc_bf  = (bfu16*)take((size_t)B * S * ENC * 2);
    bfu16* Ua_bf   = (bfu16*)take((size_t)H * ENC * 2);
    bfu16* Wih_bf  = (bfu16*)take((size_t)G3 * KX * 2);
    bfu16* Whh_bf  = (bfu16*)take((size_t)G3 * H * 2);
    bfu16* Wout_bf = (bfu16*)take((size_t)V * H * 2);
    bfu16* x_bf    = (bfu16*)take((size_t)B * KX * 2);
    float* hF0     = (float*)take((size_t)B * H * 4);
    float* hF1     = (float*)take((size_t)B * H * 4);
    bfu16* hB0     = (bfu16*)take((size_t)B * H * 2);
    bfu16* hB1     = (bfu16*)take((size_t)B * H * 2);
    float* logits  = (float*)take((size_t)B * V * 4);
    float* hF[2] = {hF0, hF1};
    bfu16* hB[2] = {hB0, hB1};

    float* out_lp = (float*)d_out;                    // (B, L, V)
    float* out_hf = out_lp + (size_t)B * L * V;       // (1, B, H)
    float* out_at = out_hf + (size_t)B * H;           // (B, L, S)

    // one-time prep: bf16 weight conversion, h0, Uk
    f2bf_kernel<<<1024, 256, 0, stream>>>(enc,  enc_bf,  B * S * ENC);
    f2bf_kernel<<<256,  256, 0, stream>>>(Ua,   Ua_bf,   H * ENC);
    f2bf_kernel<<<1024, 256, 0, stream>>>(Wih,  Wih_bf,  G3 * KX);
    f2bf_kernel<<<1024, 256, 0, stream>>>(Whh,  Whh_bf,  G3 * H);
    f2bf_kernel<<<4096, 256, 0, stream>>>(Wout, Wout_bf, V * H);
    h0_kernel<<<B, 256, 0, stream>>>(ehid, Wht, bht, hF[0], hB[0]);
    uk_kernel<<<1024, 256, 0, stream>>>(enc_bf, Ua_bf, bUa, Uk);

    for (int t = 0; t < L; ++t) {
        int rb = t & 1, wb = rb ^ 1;   // double-buffered hidden state
        attn_kernel<<<B, 256, 0, stream>>>(t, tgt, emb, Wa, ba, Va, bVa, Uk, enc,
                                           hF[rb], x_bf, out_at);
        gru_kernel<<<8, 256, 0, stream>>>(x_bf, hB[rb], hF[rb], Wih_bf, Whh_bf,
                                          bih, bhh, hF[wb], hB[wb]);
        logits_kernel<<<250, 256, 0, stream>>>(hB[wb], Wout_bf, bout, logits);
        lsm_kernel<<<B, 256, 0, stream>>>(t, logits, out_lp);
    }
    // final hidden lives in the buffer written by step L-1
    hfinal_kernel<<<(B * H + 255) / 256, 256, 0, stream>>>(hF[((L - 1) & 1) ^ 1], out_hf);
}
